// GlobalAttention_52115133169747
// MI455X (gfx1250) — matrix operand order repaired and checked by reading, not compiled, not run on hardware
//
#include <hip/hip_runtime.h>
#include <hip/hip_bf16.h>

// ---------------------------------------------------------------------------
// Bahdanau attention for MI455X (gfx1250, wave32, WMMA).
// Heavy GEMM (enc_hids @ W_enc, 137 GFLOP) runs on v_wmma_f32_16x16x32_bf16.
// ctx streaming uses GLOBAL_LOAD_ASYNC_TO_LDS double-buffering when available.
// ---------------------------------------------------------------------------

#define NB   64      // batch
#define LL   1024    // sequence length
#define HH   1024    // hidden (dec_H == enc_H_total)

typedef __attribute__((ext_vector_type(16))) __bf16 v16bf;
typedef __attribute__((ext_vector_type(8)))  float  v8f;
typedef __attribute__((ext_vector_type(4)))  int    v4i;

static __device__ __forceinline__ unsigned short f2bfbits(float x) {
  unsigned u = __float_as_uint(x);
  u += 0x7fffu + ((u >> 16) & 1u);   // round-to-nearest-even
  return (unsigned short)(u >> 16);
}

#if defined(__gfx1250__) &&                                        \
    __has_builtin(__builtin_amdgcn_global_load_async_to_lds_b128) && \
    __has_builtin(__builtin_amdgcn_s_wait_asynccnt)
#define USE_ASYNC_LDS 1
typedef __attribute__((address_space(3))) v4i lds_v4i_t;
static __device__ __forceinline__ v4i* to_glb_v4(const void* p) {
  return (v4i*)(uintptr_t)p;                       // generic ptr, const stripped
}
static __device__ __forceinline__ lds_v4i_t* to_lds_v4(void* p) {
  return reinterpret_cast<lds_v4i_t*>((unsigned)(uintptr_t)p);  // LDS offset
}
#endif

// ---------------------------------------------------------------------------
// Kernel 1: swizzle W_enc (f32 row-major [K][col]) into bf16 WMMA-B fragment
// order: [nt(64)][kt(32)][lane(32)][e(16)].
// B 32x16 bf16 order used here: lanes 0-15 hold K=0..15 (elem e == K-kbase),
// lanes 16-31 hold K=16..31; N = lane&15.  4 elems (consecutive e, same lane)
// per thread -> one b64 store.
// ---------------------------------------------------------------------------
__global__ void swz_wenc_kernel(const float* __restrict__ W,
                                unsigned short* __restrict__ o) {
  int i4   = (blockIdx.x * 256 + threadIdx.x) * 4;  // 0 .. 1M-1 step 4
  int e0   = i4 & 15;
  int lane = (i4 >> 4) & 31;
  int kt   = (i4 >> 9) & 31;
  int nt   = i4 >> 14;
  int K0   = kt * 32 + ((lane >= 16) ? 16 : 0) + e0;
  int col  = nt * 16 + (lane & 15);
  unsigned long long h0 = f2bfbits(W[(size_t)(K0 + 0) * HH + col]);
  unsigned long long h1 = f2bfbits(W[(size_t)(K0 + 1) * HH + col]);
  unsigned long long h2 = f2bfbits(W[(size_t)(K0 + 2) * HH + col]);
  unsigned long long h3 = f2bfbits(W[(size_t)(K0 + 3) * HH + col]);
  *(unsigned long long*)(o + i4) = h0 | (h1 << 16) | (h2 << 32) | (h3 << 48);
}

// ---------------------------------------------------------------------------
// Kernel 2/6: out[n,:] = (sum_s X[s-th partial])[n,:] @ W + bias.
// One block per row n; W read column-coalesced (L2 resident).
// ---------------------------------------------------------------------------
__global__ void gemm_rows_kernel(const float* __restrict__ X,
                                 const float* __restrict__ W,
                                 const float* __restrict__ bias,
                                 float* __restrict__ out,
                                 int K, int M, int nsum, int sumstride) {
  int n = blockIdx.x, tid = threadIdx.x;
  __shared__ float xs[HH];
  for (int i = tid; i < K; i += blockDim.x) {
    float s = 0.f;
    for (int ss = 0; ss < nsum; ++ss)
      s += X[(size_t)ss * sumstride + (size_t)n * K + i];
    xs[i] = s;
  }
  __syncthreads();
  for (int col = tid; col < M; col += blockDim.x) {
    float acc = bias[col];
#pragma unroll 8
    for (int k = 0; k < K; ++k)
      acc = fmaf(xs[k], W[(size_t)k * M + col], acc);
    out[(size_t)n * M + col] = acc;
  }
}

// ---------------------------------------------------------------------------
// Kernel 3: fused  scores[n,l] = sum_c tanh(d[n,c]+b_enc[c]+ (enc@W_enc)[n,l,c])
//                                * W_one[c]  + b_one
// WG = 64 threads (2 waves). WG covers 64 rows (one batch n, l-chunk of 64).
// Each wave: 2 M-tiles (32 rows), sharing each B fragment across both WMMAs.
// enc rows staged once into LDS in bf16 A-fragment order (16x32, contiguous k = 16*half + e,
// packed u64 stores); B fragments streamed from pre-swizzled W_enc (L2
// resident) as 2x global b128 per fragment.
// LDS: 4 M-tiles * 32KB (A) + 4KB (d+b_enc) = 132 KB dynamic.
// ---------------------------------------------------------------------------
__global__ __launch_bounds__(64)
void attn_scores_kernel(const float* __restrict__ enc,
                        const unsigned short* __restrict__ wswz,
                        const float* __restrict__ dws,
                        const float* __restrict__ b_enc,
                        const float* __restrict__ W_one,
                        const float* __restrict__ b_one,
                        float* __restrict__ scores) {
  extern __shared__ char smem[];                 // [4][32kt][32lane][16e] bf16 + d
  float* sD = (float*)(smem + 4 * 32768);

  const int tid = threadIdx.x;
  const int n   = blockIdx.x >> 4;
  const int l0  = (blockIdx.x & 15) * 64;

  // stage d[n,:] + b_enc
  for (int i = tid; i < HH; i += 64)
    sD[i] = dws[(size_t)n * HH + i] + b_enc[i];

  // stage 64 enc rows -> LDS, f32->bf16, A-fragment order.
  // A 16x32 bf16: lane = M + (kin&16 ? 16:0); e = kin&15.
  // Each float4 (4 consecutive K, 4-aligned) maps to one lane, consecutive e
  // -> pack to a single b64 store.
  for (int r = 0; r < 64; ++r) {
    const int mt = r >> 4, M = r & 15;
    const float4* src = (const float4*)(enc + ((size_t)(n * LL + l0 + r)) * HH);
    for (int c4 = tid; c4 < 256; c4 += 64) {
      __builtin_prefetch(src + c4 + 256, 0, 1);  // next row ahead
      float4 v = src[c4];
      int kbase = c4 * 4, kin = kbase & 31, kt = kbase >> 5;
      int lt = M + ((kin & 16) ? 16 : 0);
      int e0 = kin & 15;
      unsigned long long h0 = f2bfbits(v.x), h1 = f2bfbits(v.y);
      unsigned long long h2 = f2bfbits(v.z), h3 = f2bfbits(v.w);
      *(unsigned long long*)(smem + mt * 32768 + kt * 1024 + lt * 32 + e0 * 2) =
          h0 | (h1 << 16) | (h2 << 32) | (h3 << 48);
    }
  }
  __syncthreads();

  const int wv = tid >> 5, lane = tid & 31;
  const int col = lane & 15, half = lane >> 4;
  const char* a0base = smem + (size_t)(2 * wv) * 32768 + lane * 32;
  const char* a1base = a0base + 32768;

  float sc0[8], sc1[8];
#pragma unroll
  for (int r = 0; r < 8; ++r) { sc0[r] = 0.f; sc1[r] = 0.f; }

  for (int nt = 0; nt < 64; ++nt) {
    const float w1 = W_one[nt * 16 + col];
    const float dv = sD[nt * 16 + col];
    v8f acc0 = {}; v8f acc1 = {};
    const v16bf* bbase = (const v16bf*)(wswz + ((size_t)nt * 1024 + lane) * 16);
#pragma unroll 4
    for (int kt = 0; kt < 32; ++kt) {
      v16bf a0 = *(const v16bf*)(a0base + kt * 1024);
      v16bf a1 = *(const v16bf*)(a1base + kt * 1024);
      v16bf b  = bbase[kt * 32];
      acc0 = __builtin_amdgcn_wmma_f32_16x16x32_bf16(
          false, a0, false, b, (short)0, acc0, false, false);
      acc1 = __builtin_amdgcn_wmma_f32_16x16x32_bf16(
          false, a1, false, b, (short)0, acc1, false, false);
    }
#pragma unroll
    for (int r = 0; r < 8; ++r) {
      sc0[r] += tanhf(acc0[r] + dv) * w1;
      sc1[r] += tanhf(acc1[r] + dv) * w1;
    }
  }

  // reduce the 16 column-lanes of each half (C layout: lanes0-15 -> M=r,
  // lanes16-31 -> M=8+r)
#pragma unroll
  for (int r = 0; r < 8; ++r) {
    for (int m = 8; m >= 1; m >>= 1) {
      sc0[r] += __shfl_xor(sc0[r], m, 16);
      sc1[r] += __shfl_xor(sc1[r], m, 16);
    }
  }
  if (col == 0) {
    const float b1 = b_one[0];
    const size_t base = (size_t)n * LL + l0 + wv * 32 + half * 8;
#pragma unroll
    for (int r = 0; r < 8; ++r) {
      scores[base + r]      = sc0[r] + b1;
      scores[base + 16 + r] = sc1[r] + b1;
    }
  }
}

// ---------------------------------------------------------------------------
// Kernel 4: softmax over L per batch row; writes alpha (output #2).
// ---------------------------------------------------------------------------
__global__ void softmax_rows_kernel(const float* __restrict__ scores,
                                    float* __restrict__ alpha) {
  int n = blockIdx.x, tid = threadIdx.x;
  __shared__ float red[256];
  float4 v = ((const float4*)(scores + (size_t)n * LL))[tid];
  float m = fmaxf(fmaxf(v.x, v.y), fmaxf(v.z, v.w));
  red[tid] = m; __syncthreads();
  for (int s = 128; s > 0; s >>= 1) {
    if (tid < s) red[tid] = fmaxf(red[tid], red[tid + s]);
    __syncthreads();
  }
  m = red[0]; __syncthreads();
  v.x = expf(v.x - m); v.y = expf(v.y - m);
  v.z = expf(v.z - m); v.w = expf(v.w - m);
  red[tid] = v.x + v.y + v.z + v.w; __syncthreads();
  for (int s = 128; s > 0; s >>= 1) {
    if (tid < s) red[tid] += red[tid + s];
    __syncthreads();
  }
  float inv = 1.0f / red[0];
  v.x *= inv; v.y *= inv; v.z *= inv; v.w *= inv;
  ((float4*)(alpha + (size_t)n * LL))[tid] = v;
}

// ---------------------------------------------------------------------------
// Kernel 5: ctx partials: ctxp[s][n][e] = sum_{l in chunk s} alpha[n,l]*enc[n,l,e]
// Memory-bound streaming of enc_hids (second and last HBM read, 256 MB).
// When available, rows are fetched with GLOBAL_LOAD_ASYNC_TO_LDS_B128 double
// buffering + s_wait_asynccnt; each thread consumes only the 16 LDS bytes it
// issued itself, so no barriers are needed in the steady-state loop.
// ---------------------------------------------------------------------------
__global__ void ctx_partial_kernel(const float* __restrict__ alpha,
                                   const float* __restrict__ enc,
                                   float* __restrict__ ctxp) {
  int n = blockIdx.x, s = blockIdx.y, tid = threadIdx.x;
  __shared__ float als[128];
  if (tid < 128) als[tid] = alpha[(size_t)n * LL + s * 128 + tid];
  __syncthreads();
  float ax = 0.f, ay = 0.f, az = 0.f, aw = 0.f;
  const float* gbase = enc + ((size_t)(n * LL + s * 128)) * HH;

#if defined(USE_ASYNC_LDS)
  __shared__ float buf[2][HH];                       // 2 x 4KB double buffer
  __builtin_amdgcn_global_load_async_to_lds_b128(
      to_glb_v4(gbase + tid * 4), to_lds_v4(&buf[0][tid * 4]), 0, 0);
  for (int l = 0; l < 127; ++l) {
    const int cur = l & 1;
    __builtin_amdgcn_global_load_async_to_lds_b128(
        to_glb_v4(gbase + (size_t)(l + 1) * HH + tid * 4),
        to_lds_v4(&buf[cur ^ 1][tid * 4]), 0, 0);
    __builtin_amdgcn_s_wait_asynccnt(1);             // oldest (cur) is done
    float a = als[l];
    ax = fmaf(a, buf[cur][tid * 4 + 0], ax);
    ay = fmaf(a, buf[cur][tid * 4 + 1], ay);
    az = fmaf(a, buf[cur][tid * 4 + 2], az);
    aw = fmaf(a, buf[cur][tid * 4 + 3], aw);
  }
  __builtin_amdgcn_s_wait_asynccnt(0);
  {
    float a = als[127];
    ax = fmaf(a, buf[1][tid * 4 + 0], ax);
    ay = fmaf(a, buf[1][tid * 4 + 1], ay);
    az = fmaf(a, buf[1][tid * 4 + 2], az);
    aw = fmaf(a, buf[1][tid * 4 + 3], aw);
  }
#else
  const float4* ebase = (const float4*)gbase + tid;
  for (int l = 0; l < 128; ++l) {
    __builtin_prefetch(ebase + (l + 2) * 256, 0, 1);
    float4 v = ebase[l * 256];
    float a = als[l];
    ax = fmaf(a, v.x, ax); ay = fmaf(a, v.y, ay);
    az = fmaf(a, v.z, az); aw = fmaf(a, v.w, aw);
  }
#endif
  float4 o = {ax, ay, az, aw};
  ((float4*)(ctxp + (size_t)s * (NB * HH) + (size_t)n * HH))[tid] = o;
}

// ---------------------------------------------------------------------------
extern "C" void kernel_launch(void* const* d_in, const int* in_sizes, int n_in,
                              void* d_out, int out_size, void* d_ws, size_t ws_size,
                              hipStream_t stream) {
  const float* dec_h = (const float*)d_in[0];
  const float* enc   = (const float*)d_in[1];
  const float* W_dec = (const float*)d_in[2];
  const float* b_dec = (const float*)d_in[3];
  const float* W_enc = (const float*)d_in[4];
  const float* b_enc = (const float*)d_in[5];
  const float* W_one = (const float*)d_in[6];
  const float* b_one = (const float*)d_in[7];
  const float* W_att = (const float*)d_in[8];
  const float* b_att = (const float*)d_in[9];

  float* out_hatt  = (float*)d_out;                 // N*H
  float* out_alpha = (float*)d_out + NB * HH;       // N*L

  // workspace layout
  char* ws = (char*)d_ws;
  unsigned short* wswz = (unsigned short*)ws;                   // 2 MB bf16 W_enc
  float* dws    = (float*)(ws + (2u << 20));                    // 256 KB d
  float* scores = (float*)(ws + (2u << 20) + (1u << 18));       // 256 KB scores
  float* ctxp   = (float*)(ws + (2u << 20) + (1u << 19));       // 2 MB ctx partials

  // 1) swizzle W_enc to bf16 B-fragment order (4 elems/thread, b64 stores)
  swz_wenc_kernel<<<(HH * HH) / 1024, 256, 0, stream>>>(W_enc, wswz);

  // 2) d = dec_h @ W_dec + b_dec
  gemm_rows_kernel<<<NB, 256, 0, stream>>>(dec_h, W_dec, b_dec, dws,
                                           HH, HH, 1, 0);

  // 3) WMMA fused scores
  const int smem_bytes = 4 * 32768 + HH * (int)sizeof(float);   // 132 KB
  attn_scores_kernel<<<NB * 16, 64, smem_bytes, stream>>>(
      enc, wswz, dws, b_enc, W_one, b_one, scores);

  // 4) softmax -> alpha (output #2)
  softmax_rows_kernel<<<NB, 256, 0, stream>>>(scores, out_alpha);

  // 5) ctx partials (8-way L split)
  ctx_partial_kernel<<<dim3(NB, 8), 256, 0, stream>>>(out_alpha, enc, ctxp);

  // 6) h_att = (sum_s ctxp) @ W_att + b_att  (output #1)
  gemm_rows_kernel<<<NB, 256, 0, stream>>>(ctxp, W_att, b_att, out_hatt,
                                           HH, HH, 8, NB * HH);
}
